// Rerank_Module_28166395527480
// MI455X (gfx1250) — matrix operand order repaired and checked by reading, not compiled, not run on hardware
//
#include <hip/hip_runtime.h>

// ================= static problem config =================
#define Bz    8
#define Nq    256
#define KTOP  16
#define Cc    256
#define NH    8
#define HDim  32
#define FFd   1024
#define HPc   96
#define WPc   128
#define Ptot  12288
#define Stot  16320
#define LQ    4096      // Nq*KTOP
#define BLQ   32768     // Bz*LQ
#define BSs   130560    // Bz*Stot
#define BN    2048      // Bz*Nq

typedef __attribute__((ext_vector_type(16))) __bf16       v16bf;
typedef __attribute__((ext_vector_type(8)))  float        v8f;
typedef __attribute__((ext_vector_type(8)))  unsigned int v8u;

#if __has_builtin(__builtin_amdgcn_global_load_async_to_lds_b128) && \
    __has_builtin(__builtin_amdgcn_s_wait_asynccnt)
#define USE_ASYNC_LDS 1
// builtin parameter type: pointer to 16-byte int vector, in AS(1)/AS(3)
typedef int v4i_vec __attribute__((vector_size(16)));
typedef __attribute__((address_space(1))) v4i_vec* glb_v4i;
typedef __attribute__((address_space(3))) v4i_vec* lds_v4i;
#endif

__device__ __forceinline__ unsigned short f2bf(float f) {
  unsigned int u = __builtin_bit_cast(unsigned int, f);
  u += 0x7FFFu + ((u >> 16) & 1u);            // round-to-nearest-even
  return (unsigned short)(u >> 16);
}
__device__ __forceinline__ float bf2f(unsigned short h) {
  return __builtin_bit_cast(float, ((unsigned int)h) << 16);
}

// ================= elementwise converts =================
__global__ __launch_bounds__(256) void f2b_kernel(const float* __restrict__ in,
                                                  unsigned short* __restrict__ out, int n) {
  int i = blockIdx.x * 256 + threadIdx.x;
  if (i < n) out[i] = f2bf(in[i]);
}

// out[r, c] = in[r/div, c]   (f32, C=256); div=1 => plain copy
__global__ __launch_bounds__(256) void expand_rows_kernel(const float* __restrict__ in,
                                                          float* __restrict__ out,
                                                          int total, int div) {
  int gid = blockIdx.x * 256 + threadIdx.x;
  if (gid >= total) return;
  int c = gid & 255, r = gid >> 8;
  out[gid] = in[(size_t)(r / div) * Cc + c];
}

// concat [M,512] bf16: cols 0..255 = s0[r/div], cols 256..511 = s1[r]
__global__ __launch_bounds__(256) void concat_kernel(const float* __restrict__ s0,
                                                     const float* __restrict__ s1,
                                                     unsigned short* __restrict__ out,
                                                     int M, int div) {
  int gid = blockIdx.x * 256 + threadIdx.x;
  if (gid >= M * 512) return;
  int col = gid & 511, r = gid >> 9;
  float v = (col < 256) ? s0[(size_t)(r / div) * Cc + col]
                        : s1[(size_t)r * Cc + (col - 256)];
  out[gid] = f2bf(v);
}

// ================= top-k (block per (b,n) row) =================
__global__ __launch_bounds__(256) void topk_kernel(const float* __restrict__ ct,
                                                   float* __restrict__ tnorm,
                                                   float* __restrict__ locs) {
  __shared__ float sval[256];
  __shared__ int   sidx[256];
  __shared__ int   chosen[KTOP];
  const int row = blockIdx.x;              // b*Nq + n
  const int t = threadIdx.x;
  const float* rp = ct + (size_t)row * Ptot;
  for (int pass = 0; pass < KTOP; ++pass) {
    float bv = -3.4e38f; int bi = 1 << 30;
    for (int i = t; i < Ptot; i += 256) {
      bool skip = false;
      for (int j = 0; j < pass; ++j) skip |= (chosen[j] == i);
      if (skip) continue;
      float v = rp[i];
      if (v > bv || (v == bv && i < bi)) { bv = v; bi = i; }
    }
    sval[t] = bv; sidx[t] = bi;
    __syncthreads();
    for (int s = 128; s > 0; s >>= 1) {
      if (t < s) {
        float ov = sval[t + s]; int oi = sidx[t + s];
        if (ov > sval[t] || (ov == sval[t] && oi < sidx[t])) { sval[t] = ov; sidx[t] = oi; }
      }
      __syncthreads();
    }
    if (t == 0) {
      int ii = sidx[0];
      chosen[pass] = ii;
      float col = (float)(ii % WPc), rw = (float)(ii / WPc);
      float x = (col + 0.5f) * 4.f, y = (rw + 0.5f) * 4.f;
      size_t o = ((size_t)row * KTOP + pass) * 2;
      locs[o] = x; locs[o + 1] = y;
      float tx = fminf(fmaxf(x / 512.f, 0.f), 1.f);
      float ty = fminf(fmaxf(y / 384.f, 0.f), 1.f);
      tnorm[o] = tx; tnorm[o + 1] = ty;
    }
    __syncthreads();
  }
}

// ================= feature pyramid (writes bf16 f_scales) =================
__global__ __launch_bounds__(256) void lvl0_kernel(const float* __restrict__ h,
                                                   unsigned short* __restrict__ fsc) {
  int gid = blockIdx.x * 256 + threadIdx.x;
  if (gid >= Bz * Ptot * Cc) return;
  int c = gid & 255, t = gid >> 8;
  int p = t % Ptot, b = t / Ptot;
  fsc[((size_t)b * Stot + p) * Cc + c] = f2bf(h[gid]);
}
__global__ __launch_bounds__(256) void pool2_kernel(const float* __restrict__ h,
                                                    unsigned short* __restrict__ fsc) {
  int gid = blockIdx.x * 256 + threadIdx.x;
  if (gid >= Bz * 48 * 64 * Cc) return;
  int c = gid & 255, t = gid >> 8;
  int ow = t % 64; t /= 64; int oh = t % 48; int b = t / 48;
  const float* hb = h + (size_t)b * Ptot * Cc;
  float s = 0.f;
  for (int dy = 0; dy < 2; ++dy)
    for (int dx = 0; dx < 2; ++dx)
      s += hb[((size_t)((2 * oh + dy) * WPc + 2 * ow + dx)) * Cc + c];
  fsc[((size_t)b * Stot + 12288 + oh * 64 + ow) * Cc + c] = f2bf(s * 0.25f);
}
__global__ __launch_bounds__(256) void pool4_kernel(const float* __restrict__ h,
                                                    unsigned short* __restrict__ fsc) {
  int gid = blockIdx.x * 256 + threadIdx.x;
  if (gid >= Bz * 24 * 32 * Cc) return;
  int c = gid & 255, t = gid >> 8;
  int ow = t % 32; t /= 32; int oh = t % 24; int b = t / 24;
  const float* hb = h + (size_t)b * Ptot * Cc;
  float s = 0.f;
  for (int dy = 0; dy < 4; ++dy)
    for (int dx = 0; dx < 4; ++dx)
      s += hb[((size_t)((4 * oh + dy) * WPc + 4 * ow + dx)) * Cc + c];
  fsc[((size_t)b * Stot + 15360 + oh * 32 + ow) * Cc + c] = f2bf(s * 0.0625f);
}
// exact replica of reference's level-4 "flattened" avg pool
__global__ __launch_bounds__(256) void poolf4_kernel(const float* __restrict__ h,
                                                     unsigned short* __restrict__ fsc) {
  int gid = blockIdx.x * 256 + threadIdx.x;
  if (gid >= Bz * 192 * Cc) return;
  int r = gid & 255, t = gid >> 8;
  int j = t % 192, b = t / 192;
  int lin = r * 192 + j;
  int i = lin >> 5, kk = lin & 31;
  const float* hb = h + (size_t)b * Ptot * Cc;
  float s = 0.f;
  for (int u = 0; u < 8; ++u)
    for (int v = 0; v < 8; ++v)
      s += hb[(size_t)(((i * 8 + u) * 32 + kk) * 8 + v)];
  fsc[((size_t)b * Stot + 16128 + j) * Cc + r] = f2bf(s * (1.f / 64.f));
}

// ================= bf16 WMMA GEMM: out = A[M,K] * B[K,N] (+bias,+resid,relu) =================
// M%128==0, N%64==0, K%32==0. Block: 256 thr (8 waves), 128x64 tile, K step 32.
// Wave w covers rows w*16..w*16+15 and all 64 columns (4 WMMA accumulators),
// so one A fragment feeds 4 v_wmma ops per K step.
__global__ __launch_bounds__(256) void gemm_bf16_kernel(
    const unsigned short* __restrict__ A,   // [M,K] bf16 row-major
    const unsigned short* __restrict__ Bm,  // [K,N] bf16 row-major
    const float* __restrict__ bias,         // [N] or null
    const float* __restrict__ resid,        // [M,N] f32 or null
    float* __restrict__ outF,               // f32 out, or null
    unsigned short* __restrict__ outB,      // bf16 out if outF==null
    int M, int N, int Kd, int relu) {
  __shared__ unsigned short As[128 * 32];   // [128][32] row-major (8 KB)
  __shared__ unsigned short BsT[64 * 34];   // [n(64)][k(32)+pad]
  const int tid = threadIdx.x;
  const int wave = tid >> 5, lane = tid & 31;
  const int l16 = lane & 15, hi = lane >> 4;
  const int m0 = blockIdx.x * 128, n0 = blockIdx.y * 64;
  const int wm = wave * 16;                 // 8 row-waves cover 128 rows
  const v8f vzero = {0.f, 0.f, 0.f, 0.f, 0.f, 0.f, 0.f, 0.f};
  v8f acc[4] = {vzero, vzero, vzero, vzero};
  const int ar = tid >> 2, ap = tid & 3;    // A tile: rows ar, ar+64; 8-elem part ap
  const int bk = tid >> 3, bp = tid & 7;    // B tile: k row, 8-col part
  for (int k0 = 0; k0 < Kd; k0 += 32) {
    // ---- stage A tile (global -> LDS). Async DMA path when available. ----
    const unsigned short* ga0 = A + (size_t)(m0 + ar) * Kd + (k0 + ap * 8);
    const unsigned short* ga1 = A + (size_t)(m0 + ar + 64) * Kd + (k0 + ap * 8);
    unsigned short* la0 = &As[ar * 32 + ap * 8];
    unsigned short* la1 = &As[(ar + 64) * 32 + ap * 8];
#if defined(USE_ASYNC_LDS)
    __builtin_amdgcn_global_load_async_to_lds_b128((glb_v4i)ga0, (lds_v4i)la0, 0, 0);
    __builtin_amdgcn_global_load_async_to_lds_b128((glb_v4i)ga1, (lds_v4i)la1, 0, 0);
#else
    *(uint4*)la0 = *(const uint4*)ga0;
    *(uint4*)la1 = *(const uint4*)ga1;
#endif
    // ---- stage B tile transposed (needs per-element shuffle, keep sync) ----
    uint4 bv = *(const uint4*)(Bm + (size_t)(k0 + bk) * N + (n0 + bp * 8));
    const unsigned short* bs = (const unsigned short*)&bv;
#pragma unroll
    for (int e = 0; e < 8; ++e) BsT[(bp * 8 + e) * 34 + bk] = bs[e];
    if (k0 + 32 < Kd) {   // uniform branch; hint next K tile toward the caches
      __builtin_prefetch(A + (size_t)(m0 + ar) * Kd + (k0 + 32 + ap * 8), 0, 1);
      __builtin_prefetch(Bm + (size_t)(k0 + 32 + bk) * N + (n0 + bp * 8), 0, 1);
    }
#if defined(USE_ASYNC_LDS)
    __builtin_amdgcn_s_wait_asynccnt(0);
#endif
    __syncthreads();
    // A fragment: lane<16 K{0..15}, lane>=16 K{16..31} (matches B fragment order)
    v8u au;
    {
      const int row = wm + l16, kb = hi * 16;
#pragma unroll
      for (int j = 0; j < 8; ++j) {
        const int kk = kb + 2 * j;
        au[j] = *(const unsigned int*)(&As[row * 32 + kk]);
      }
    }
    const v16bf afrag = __builtin_bit_cast(v16bf, au);
    // B fragments: lane l16 = column, hi selects K half 0..15 / 16..31
    const int kb2 = hi * 16;
#pragma unroll
    for (int ct = 0; ct < 4; ++ct) {
      v8u bu;
      const int cb = (ct * 16 + l16) * 34 + kb2;
#pragma unroll
      for (int j = 0; j < 8; ++j)
        bu[j] = *(const unsigned int*)(&BsT[cb + 2 * j]);
      const v16bf bfrag = __builtin_bit_cast(v16bf, bu);
      acc[ct] = __builtin_amdgcn_wmma_f32_16x16x32_bf16(false, afrag, false, bfrag,
                                                        (short)0, acc[ct], false, false);
    }
    __syncthreads();
  }
  // Epilogue. C/D layout: VGPR v, lanes<16 -> M=v,N=l16 ; lanes>=16 -> M=v+8
  const int mb = m0 + wm + 8 * hi;
#pragma unroll
  for (int ct = 0; ct < 4; ++ct) {
    const int n = n0 + ct * 16 + l16;
#pragma unroll
    for (int v = 0; v < 8; ++v) {
      float x = acc[ct][v];
      const int m = mb + v;
      if (bias)  x += bias[n];
      if (resid) x += resid[(size_t)m * N + n];
      if (relu)  x = fmaxf(x, 0.f);
      const size_t oi = (size_t)m * N + n;
      if (outF) outF[oi] = x;
      else      outB[oi] = f2bf(x);
    }
  }
}

// ================= LayerNorm (block per row, C=256) =================
__global__ __launch_bounds__(256) void ln_kernel(const float* __restrict__ in,
                                                 const float* __restrict__ g,
                                                 const float* __restrict__ bt,
                                                 float* __restrict__ out) {
  __shared__ float red[256];
  const int row = blockIdx.x, t = threadIdx.x;
  const size_t base = (size_t)row * Cc;
  float x = in[base + t];
  red[t] = x; __syncthreads();
  for (int s = 128; s > 0; s >>= 1) { if (t < s) red[t] += red[t + s]; __syncthreads(); }
  float mean = red[0] * (1.f / 256.f);
  __syncthreads();
  float d = x - mean;
  red[t] = d * d; __syncthreads();
  for (int s = 128; s > 0; s >>= 1) { if (t < s) red[t] += red[t + s]; __syncthreads(); }
  float var = red[0] * (1.f / 256.f);
  out[base + t] = d * rsqrtf(var + 1e-5f) * g[t] + bt[t];
}

// ================= softmax over 16 (deformable attention weights) =================
__global__ __launch_bounds__(256) void softmax16_kernel(float* __restrict__ aw, int nrh) {
  int t = blockIdx.x * 256 + threadIdx.x;
  if (t >= nrh) return;
  int row = t >> 3, h = t & 7;
  float* p = aw + (size_t)row * 128 + h * 16;
  float m = -3.4e38f;
  for (int j = 0; j < 16; ++j) m = fmaxf(m, p[j]);
  float s = 0.f;
  for (int j = 0; j < 16; ++j) { float e = __expf(p[j] - m); p[j] = e; s += e; }
  float inv = 1.f / s;
  for (int j = 0; j < 16; ++j) p[j] *= inv;
}

// ================= deformable bilinear sampling =================
__global__ __launch_bounds__(256) void sample_kernel(const unsigned short* __restrict__ value,
                                                     const float* __restrict__ off,
                                                     const float* __restrict__ aw,
                                                     const float* __restrict__ tnorm,
                                                     unsigned short* __restrict__ samp) {
  int gid = blockIdx.x * 256 + threadIdx.x;
  if (gid >= BLQ * Cc) return;
  const int d = gid & 31;
  const int h = (gid >> 5) & 7;
  const int lqg = gid >> 8;
  const int b = lqg >> 12;
  const float rx = tnorm[(size_t)lqg * 2 + 0];
  const float ry = tnorm[(size_t)lqg * 2 + 1];
  const int Hls[4] = {96, 48, 24, 12};
  const int Wls[4] = {128, 64, 32, 16};
  const int S0s[4] = {0, 12288, 15360, 16128};
  const float* offp = off + (size_t)lqg * 256 + h * 32;
  const float* awp  = aw  + (size_t)lqg * 128 + h * 16;
  float acc = 0.f;
#pragma unroll
  for (int l = 0; l < 4; ++l) {
    const int Hl = Hls[l], Wl = Wls[l];
    const unsigned short* vb = value + ((size_t)b * Stot + S0s[l]) * Cc + h * 32 + d;
    auto corner = [&](int xi, int yi, float wgt) -> float {
      bool valid = (xi >= 0) && (xi < Wl) && (yi >= 0) && (yi < Hl);
      int xc = xi < 0 ? 0 : (xi > Wl - 1 ? Wl - 1 : xi);
      int yc = yi < 0 ? 0 : (yi > Hl - 1 ? Hl - 1 : yi);
      float v = bf2f(vb[(size_t)(yc * Wl + xc) * Cc]);
      return valid ? wgt * v : 0.f;
    };
#pragma unroll
    for (int pt = 0; pt < 4; ++pt) {
      float ox = offp[l * 8 + pt * 2], oy = offp[l * 8 + pt * 2 + 1];
      float w = awp[l * 4 + pt];
      float x = (rx + ox / (float)Wl) * (float)Wl - 0.5f;
      float y = (ry + oy / (float)Hl) * (float)Hl - 0.5f;
      float xf = floorf(x), yf = floorf(y);
      int x0 = (int)xf, y0 = (int)yf;
      float wx = x - xf, wy = y - yf;
      acc += w * (corner(x0,     y0,     (1.f - wx) * (1.f - wy)) +
                  corner(x0 + 1, y0,     wx * (1.f - wy)) +
                  corner(x0,     y0 + 1, (1.f - wx) * wy) +
                  corner(x0 + 1, y0 + 1, wx * wy));
    }
  }
  samp[gid] = f2bf(acc);
}

// ================= uncertainty head: dot(qtp[row], Wunc) + bunc =================
__global__ __launch_bounds__(256) void unc_kernel(const float* __restrict__ qtp,
                                                  const float* __restrict__ Wu,
                                                  const float* __restrict__ bu,
                                                  float* __restrict__ out) {
  int r = blockIdx.x * 256 + threadIdx.x;
  if (r >= BLQ) return;
  float a = bu[0];
  const float* p = qtp + (size_t)r * Cc;
  for (int c = 0; c < Cc; ++c) a += p[c] * Wu[c];
  out[r] = a;
}

// ================= decoder attention: Lq=1, Lk=16, per (bn,h) =================
__global__ __launch_bounds__(256) void dec_attn_kernel(const float* __restrict__ qh,
                                                       const float* __restrict__ kh,
                                                       const float* __restrict__ vh,
                                                       float* __restrict__ o) {
  int t = blockIdx.x * 256 + threadIdx.x;
  if (t >= BN * NH) return;
  const int bn = t >> 3, h = t & 7;
  const float* qp = qh + (size_t)bn * Cc + h * HDim;
  float sc[16];
  float m = -3.4e38f;
  for (int j = 0; j < 16; ++j) {
    const float* kp = kh + (size_t)(bn * 16 + j) * Cc + h * HDim;
    float dot = 0.f;
    for (int dd = 0; dd < HDim; ++dd) dot += qp[dd] * kp[dd];
    sc[j] = dot * 0.17677669529663687f;   // 1/sqrt(32)
    m = fmaxf(m, sc[j]);
  }
  float s = 0.f;
  for (int j = 0; j < 16; ++j) { sc[j] = __expf(sc[j] - m); s += sc[j]; }
  float inv = 1.f / s;
  for (int j = 0; j < 16; ++j) sc[j] *= inv;
  float* op = o + (size_t)bn * Cc + h * HDim;
  for (int dd = 0; dd < HDim; ++dd) {
    float acc = 0.f;
    for (int j = 0; j < 16; ++j)
      acc += sc[j] * vh[(size_t)(bn * 16 + j) * Cc + h * HDim + dd];
    op[dd] = acc;
  }
}

// ================= host-side orchestration =================
static inline int cdiv(int a, int b) { return (a + b - 1) / b; }

extern "C" void kernel_launch(void* const* d_in, const int* in_sizes, int n_in,
                              void* d_out, int out_size, void* d_ws, size_t ws_size,
                              hipStream_t stream) {
  (void)in_sizes; (void)n_in; (void)out_size; (void)ws_size;
  const float* q_t = (const float*)d_in[0];
  const float* h_t = (const float*)d_in[1];
  const float* c_t = (const float*)d_in[2];
  // params flattened as a JAX pytree (dict keys sorted):
  // Wfus, Wproj, Wunc, bfus, bproj, bunc, dec[3]{16}, enc[3]{16}
  const float* Wfus  = (const float*)d_in[3];
  const float* Wproj = (const float*)d_in[4];
  const float* Wunc  = (const float*)d_in[5];
  const float* bfus  = (const float*)d_in[6];
  const float* bproj = (const float*)d_in[7];
  const float* bunc  = (const float*)d_in[8];
  // dec keys sorted: W1 W2 Wk Wout Wq Wv b1 b2 bf1 bf2 bk bout bq bv g1 g2
  auto decP = [&](int L, int k) { return (const float*)d_in[9 + 16 * L + k]; };
  // enc keys sorted: W1 W2 Wa Wo Wout Wv b1 b2 ba bf1 bf2 bo bout bv g1 g2
  auto encP = [&](int L, int k) { return (const float*)d_in[57 + 16 * L + k]; };

  float* out_q    = (float*)d_out;                 // [8,256,256]
  float* out_unc  = out_q + (size_t)BN * Cc;       // [8,256,16]
  float* out_locs = out_unc + (size_t)BN * KTOP;   // [8,256,16,2]

  // ---- workspace slabs (fixed deterministic layout) ----
  char* w = (char*)d_ws;
  size_t off = 0;
  auto alloc = [&](size_t bytes) -> void* {
    void* p = w + off;
    off += (bytes + 255) & ~(size_t)255;
    return p;
  };
  float*          tnorm = (float*)alloc((size_t)BN * KTOP * 2 * 4);
  unsigned short* fsc   = (unsigned short*)alloc((size_t)BSs * Cc * 2);   // f_scales bf16
  float*          q32   = (float*)alloc((size_t)BLQ * Cc * 4);            // q_local / dec vh
  unsigned short* qb16  = (unsigned short*)alloc((size_t)BLQ * Cc * 2);
  unsigned short* bigB  = (unsigned short*)alloc((size_t)BLQ * FFd * 2);  // value/h1/concat bf16
  float*          offb  = (float*)alloc((size_t)BLQ * Cc * 4);            // offsets / dec kh
  float*          awb   = (float*)alloc((size_t)BLQ * 128 * 4);
  unsigned short* sampb = (unsigned short*)alloc((size_t)BLQ * Cc * 2);   // samp / kv bf16
  float*          tmp32 = (float*)alloc((size_t)BLQ * Cc * 4);            // pre-LN / qtp
  unsigned short* wb    = (unsigned short*)alloc((size_t)512 * 1024 * 2); // weight bf16
  float*          dq32  = (float*)alloc((size_t)BN * Cc * 4);             // decoder qq
  unsigned short* dqb   = (unsigned short*)alloc((size_t)BN * Cc * 2);
  float*          dqh   = (float*)alloc((size_t)BN * Cc * 4);
  float*          dof   = (float*)alloc((size_t)BN * Cc * 4);
  unsigned short* dob16 = (unsigned short*)alloc((size_t)BN * Cc * 2);
  float*          dt    = (float*)alloc((size_t)BN * Cc * 4);

  auto convW = [&](const float* src, int n) {
    f2b_kernel<<<cdiv(n, 256), 256, 0, stream>>>(src, wb, n);
  };
  auto gemm = [&](const void* A, const float* bias, const float* res,
                  float* oF, void* oB, int M, int Ncols, int Kd, int relu) {
    dim3 g(M / 128, Ncols / 64);
    gemm_bf16_kernel<<<g, 256, 0, stream>>>((const unsigned short*)A, wb, bias, res,
                                            oF, (unsigned short*)oB, M, Ncols, Kd, relu);
  };

  // ---- phase 0: top-k regions -> tnorm + target_locs ----
  topk_kernel<<<BN, 256, 0, stream>>>(c_t, tnorm, out_locs);

  // ---- phase 1: feature pyramid (bf16 f_scales) ----
  lvl0_kernel<<<cdiv(Bz * Ptot * Cc, 256), 256, 0, stream>>>(h_t, fsc);
  pool2_kernel<<<cdiv(Bz * 48 * 64 * Cc, 256), 256, 0, stream>>>(h_t, fsc);
  pool4_kernel<<<cdiv(Bz * 24 * 32 * Cc, 256), 256, 0, stream>>>(h_t, fsc);
  poolf4_kernel<<<cdiv(Bz * 192 * Cc, 256), 256, 0, stream>>>(h_t, fsc);

  // ---- q_local = broadcast(q_t) over K ----
  expand_rows_kernel<<<cdiv(BLQ * Cc, 256), 256, 0, stream>>>(q_t, q32, BLQ * Cc, KTOP);

  // ---- encoder: 3 deformable attention blocks ----
  for (int L = 0; L < 3; ++L) {
    const float *W1 = encP(L, 0), *W2 = encP(L, 1), *Wa = encP(L, 2), *Wo = encP(L, 3);
    const float *Wout = encP(L, 4), *Wv = encP(L, 5), *b1 = encP(L, 6), *b2 = encP(L, 7);
    const float *ba = encP(L, 8), *bf1 = encP(L, 9), *bf2 = encP(L, 10), *bo = encP(L, 11);
    const float *bout = encP(L, 12), *bv = encP(L, 13), *g1 = encP(L, 14), *g2 = encP(L, 15);

    f2b_kernel<<<cdiv(BLQ * Cc, 256), 256, 0, stream>>>(q32, qb16, BLQ * Cc);
    convW(Wv, Cc * Cc);
    gemm(fsc, bv, nullptr, nullptr, bigB, BSs, Cc, Cc, 0);            // value (bf16)
    convW(Wo, Cc * Cc);
    gemm(qb16, bo, nullptr, offb, nullptr, BLQ, Cc, Cc, 0);           // sampling offsets
    convW(Wa, Cc * 128);
    gemm(qb16, ba, nullptr, awb, nullptr, BLQ, 128, Cc, 0);           // attn weights
    softmax16_kernel<<<cdiv(BLQ * NH, 256), 256, 0, stream>>>(awb, BLQ * NH);
    sample_kernel<<<cdiv(BLQ * Cc, 256), 256, 0, stream>>>(bigB, offb, awb, tnorm, sampb);
    convW(Wout, Cc * Cc);
    gemm(sampb, bout, q32, tmp32, nullptr, BLQ, Cc, Cc, 0);           // out proj + residual
    ln_kernel<<<BLQ, 256, 0, stream>>>(tmp32, g1, b1, q32);
    f2b_kernel<<<cdiv(BLQ * Cc, 256), 256, 0, stream>>>(q32, qb16, BLQ * Cc);
    convW(W1, Cc * FFd);
    gemm(qb16, bf1, nullptr, nullptr, bigB, BLQ, FFd, Cc, 1);         // FFN up + relu (bf16)
    convW(W2, FFd * Cc);
    gemm(bigB, bf2, q32, tmp32, nullptr, BLQ, Cc, FFd, 0);            // FFN down + residual
    ln_kernel<<<BLQ, 256, 0, stream>>>(tmp32, g2, b2, q32);
  }

  // ---- qtp = concat(qe, top_patches) @ Wproj + bproj ; unc head ----
  concat_kernel<<<cdiv(BLQ * 512, 256), 256, 0, stream>>>(q_t, q32, bigB, BLQ, KTOP);
  convW(Wproj, 512 * Cc);
  gemm(bigB, bproj, nullptr, tmp32, nullptr, BLQ, Cc, 512, 0);        // qtp (f32)
  unc_kernel<<<cdiv(BLQ, 256), 256, 0, stream>>>(tmp32, Wunc, bunc, out_unc);
  f2b_kernel<<<cdiv(BLQ * Cc, 256), 256, 0, stream>>>(tmp32, sampb, BLQ * Cc);  // kv bf16
  expand_rows_kernel<<<cdiv(BN * Cc, 256), 256, 0, stream>>>(q_t, dq32, BN * Cc, 1);

  // ---- decoder: 3 MHA blocks (Lq=1, Lk=16, per (b,n)) ----
  for (int L = 0; L < 3; ++L) {
    const float *W1 = decP(L, 0), *W2 = decP(L, 1), *Wk = decP(L, 2), *Wout = decP(L, 3);
    const float *Wq = decP(L, 4), *Wv = decP(L, 5), *b1 = decP(L, 6), *b2 = decP(L, 7);
    const float *bf1 = decP(L, 8), *bf2 = decP(L, 9), *bk = decP(L, 10), *bout = decP(L, 11);
    const float *bq = decP(L, 12), *bv = decP(L, 13), *g1 = decP(L, 14), *g2 = decP(L, 15);

    f2b_kernel<<<cdiv(BN * Cc, 256), 256, 0, stream>>>(dq32, dqb, BN * Cc);
    convW(Wq, Cc * Cc);
    gemm(dqb, bq, nullptr, dqh, nullptr, BN, Cc, Cc, 0);              // qh
    convW(Wk, Cc * Cc);
    gemm(sampb, bk, nullptr, offb, nullptr, BLQ, Cc, Cc, 0);          // kh (reuse slab)
    convW(Wv, Cc * Cc);
    gemm(sampb, bv, nullptr, q32, nullptr, BLQ, Cc, Cc, 0);           // vh (reuse slab)
    dec_attn_kernel<<<cdiv(BN * NH, 256), 256, 0, stream>>>(dqh, offb, q32, dof);
    f2b_kernel<<<cdiv(BN * Cc, 256), 256, 0, stream>>>(dof, dob16, BN * Cc);
    convW(Wout, Cc * Cc);
    gemm(dob16, bout, dq32, dt, nullptr, BN, Cc, Cc, 0);
    ln_kernel<<<BN, 256, 0, stream>>>(dt, g1, b1, dq32);
    f2b_kernel<<<cdiv(BN * Cc, 256), 256, 0, stream>>>(dq32, dqb, BN * Cc);
    convW(W1, Cc * FFd);
    gemm(dqb, bf1, nullptr, nullptr, bigB, BN, FFd, Cc, 1);
    convW(W2, FFd * Cc);
    gemm(bigB, bf2, dq32, dt, nullptr, BN, Cc, FFd, 0);
    ln_kernel<<<BN, 256, 0, stream>>>(dt, g2, b2, dq32);
  }

  // ---- fusion: q_out = concat(q_t, qq) @ Wfus + bfus -> d_out ----
  concat_kernel<<<cdiv(BN * 512, 256), 256, 0, stream>>>(q_t, dq32, bigB, BN, 1);
  convW(Wfus, 512 * Cc);
  gemm(bigB, bfus, nullptr, out_q, nullptr, BN, Cc, 512, 0);
}
